// EquiConv_9972914061333
// MI455X (gfx1250) — compile-verified, matrix operand order checked by reading, not run on hardware
//
#include <hip/hip_runtime.h>
#include <hip/hip_bf16.h>

typedef __bf16 bf16_t;
typedef __attribute__((ext_vector_type(16))) __bf16 v16bf;
typedef __attribute__((ext_vector_type(8)))  __bf16 v8bf;
typedef __attribute__((ext_vector_type(8)))  float  v8f;

#define E_TOT   20000
#define ROW_IN  160      // S + 3V
#define FCIN    128
#define MB      32       // edges per block (2 x 16-row M tiles)
#define INV_SQRT3 0.57735026918962576f

// fragment-region bases (in fragment units; 1 fragment = 512 bf16 = 1KB)
#define FR_SS   0        // K=4096 (u*64+v), N=96   -> 128*6 = 768 frags
#define FR_VV   768      // K=1024 (u*32+v), N=96   ->  32*6 = 192 frags
#define FR_SV   960      // K=2048 (u*32+v), N=32   ->  64*2 = 128 frags
#define FR_VS   1088     // K=2048 (u*64+v), N=32   ->  64*2 = 128 frags
#define FR_FC1  1216     // K=128, N=64             ->   4*4 =  16 frags
#define FR_FC2  1232     // K=64,  N=64             ->   2*4 =   8 frags
#define FR_FC3  1240     // K=64,  N=96             ->   2*6 =  12 frags
#define FR_TOT  1252

// LDS row pitches (elems): keep every v8bf chunk 16B aligned, rows bank-spread
#define P64  72
#define P32  40
#define PFW  136
#define PH   72
#define PSC  100
#define PVC  36
#define PW   100

// ---------------------------------------------------------------------------
// Weight prepack: fp32 -> bf16 per-lane WMMA B fragments (32B/lane/fragment).
//   lane<16 : N=lane,    K = {klo..klo+7, 16+klo..16+klo+7}, klo=0
//   lane>=16: N=lane-16, klo=8
// ---------------------------------------------------------------------------
__global__ void equiconv_prepack(const float* __restrict__ wss_s,
                                 const float* __restrict__ wvv_s,
                                 const float* __restrict__ wss_g,
                                 const float* __restrict__ wvv_g,
                                 const float* __restrict__ wsv,
                                 const float* __restrict__ wvs,
                                 const float* __restrict__ fc1,
                                 const float* __restrict__ fc2,
                                 const float* __restrict__ fc3,
                                 bf16_t* __restrict__ out) {
  int gid = blockIdx.x * blockDim.x + threadIdx.x;
  if (gid >= FR_TOT * 32) return;
  int frag = gid >> 5;
  int lane = gid & 31;
  int klo  = (lane < 16) ? 0 : 8;
  int ncol = lane & 15;
  bf16_t* dst = out + (size_t)frag * 512 + lane * 16;
  #pragma unroll
  for (int t = 0; t < 16; ++t) {
    int kk = (t < 8) ? (klo + t) : (16 + klo + (t - 8));
    float val;
    if (frag < FR_VV) {                         // SS: k = u*64+v
      int f = frag, kt = f / 6, nt = f % 6;
      int kg = kt * 32 + kk, u = kg >> 6, v = kg & 63;
      int n = nt * 16 + ncol;
      val = (n < 64) ? wss_s[(u * 64 + v) * 64 + n]
                     : wss_g[(u * 64 + v) * 32 + (n - 64)];
    } else if (frag < FR_SV) {                  // VV: k = u*32+v, pre-scaled
      int f = frag - FR_VV, kt = f / 6, nt = f % 6;
      int kg = kt * 32 + kk, u = kg >> 5, v = kg & 31;
      int n = nt * 16 + ncol;
      val = INV_SQRT3 * ((n < 64) ? wvv_s[(u * 32 + v) * 64 + n]
                                  : wvv_g[(u * 32 + v) * 32 + (n - 64)]);
    } else if (frag < FR_VS) {                  // SV: k = u*32+v
      int f = frag - FR_SV, kt = f / 2, nt = f % 2;
      int kg = kt * 32 + kk, u = kg >> 5, v = kg & 31;
      val = wsv[(u * 32 + v) * 32 + nt * 16 + ncol];
    } else if (frag < FR_FC1) {                 // VS: k = u*64+v
      int f = frag - FR_VS, kt = f / 2, nt = f % 2;
      int kg = kt * 32 + kk, u = kg >> 6, v = kg & 63;
      val = wvs[(u * 64 + v) * 32 + nt * 16 + ncol];
    } else if (frag < FR_FC2) {
      int f = frag - FR_FC1, kt = f / 4, nt = f % 4;
      val = fc1[(kt * 32 + kk) * 64 + nt * 16 + ncol];
    } else if (frag < FR_FC3) {
      int f = frag - FR_FC2, kt = f / 4, nt = f % 4;
      val = fc2[(kt * 32 + kk) * 64 + nt * 16 + ncol];
    } else {
      int f = frag - FR_FC3, kt = f / 6, nt = f % 6;
      val = fc3[(kt * 32 + kk) * 96 + nt * 16 + ncol];
    }
    dst[t] = (bf16_t)val;
  }
}

// ---------------------------------------------------------------------------
__device__ __forceinline__ v16bf load_a(const bf16_t* row, int kbase, int klo) {
  v8bf lo = *(const v8bf*)(row + kbase + klo);        // ds_load_b128
  v8bf hi = *(const v8bf*)(row + kbase + 16 + klo);   // ds_load_b128
  return __builtin_shufflevector(lo, hi, 0,1,2,3,4,5,6,7,8,9,10,11,12,13,14,15);
}

__device__ __forceinline__ v16bf load_b(const bf16_t* wp, int frag, int lane) {
  return *(const v16bf*)(wp + (size_t)frag * 512 + lane * 16); // 2x global_load_b128
}

__device__ __forceinline__ v8f wmma_bf16(v16bf a, v16bf b, v8f c) {
  return __builtin_amdgcn_wmma_f32_16x16x32_bf16(false, a, false, b,
                                                 (short)0, c, false, false);
}

__device__ __forceinline__ float silu_f(float x) {
  return x / (1.0f + __expf(-x));
}

// ---------------------------------------------------------------------------
// Main kernel: 32 edges per block (2 M-tiles), 512 threads = 16 waves.
// Every wave keeps two accumulators and reuses each B fragment for both
// M-tiles (halves L2 B traffic per flop). K loops unrolled for deeper
// load pipelining (bigger s_clause groups, partial loadcnt waits).
// ---------------------------------------------------------------------------
__launch_bounds__(512)
__global__ void equiconv_main(const float* __restrict__ fea1,
                              const float* __restrict__ fea2,
                              const float* __restrict__ feaw,
                              const float* __restrict__ fcb1,
                              const float* __restrict__ fcb2,
                              const float* __restrict__ fcb3,
                              const bf16_t* __restrict__ wp,
                              float* __restrict__ out) {
  __shared__ __align__(16) bf16_t x1s[MB * P64];
  __shared__ __align__(16) bf16_t x2s[MB * P64];
  __shared__ __align__(16) bf16_t x1v[3][MB * P32];
  __shared__ __align__(16) bf16_t x2v[3][MB * P32];
  __shared__ __align__(16) bf16_t fw [MB * PFW];
  __shared__ __align__(16) bf16_t hb [MB * PH];
  __shared__ __align__(16) float  scg[MB * PSC];     // sc (0..63) | g (64..95)
  __shared__ __align__(16) float  vecc[3][MB * PVC]; // vec per i
  __shared__ __align__(16) float  wb [MB * PW];      // MLP output w (96)

  const int tid   = threadIdx.x;
  const int ebase = blockIdx.x * MB;

  // ---- stage inputs to LDS as bf16 ----
  for (int idx = tid; idx < MB * ROW_IN; idx += 512) {
    int m = idx / ROW_IN, c = idx % ROW_IN;
    float f1 = fea1[(size_t)(ebase + m) * ROW_IN + c];
    float f2 = fea2[(size_t)(ebase + m) * ROW_IN + c];
    if (c < 64) {
      x1s[m * P64 + c] = (bf16_t)f1;
      x2s[m * P64 + c] = (bf16_t)f2;
    } else {
      int c2 = c - 64, u = c2 / 3, i = c2 % 3;   // reshape (V,3)
      x1v[i][m * P32 + u] = (bf16_t)f1;
      x2v[i][m * P32 + u] = (bf16_t)f2;
    }
  }
  for (int idx = tid; idx < MB * FCIN; idx += 512) {
    int m = idx >> 7, c = idx & 127;
    fw[m * PFW + c] = (bf16_t)feaw[(size_t)(ebase + m) * FCIN + c];
  }
  __syncthreads();

  const int wave  = tid >> 5, lane = tid & 31;
  const int m0    = lane & 15;        // M-tile 0 row
  const int m1    = m0 + 16;          // M-tile 1 row
  const int klo   = (lane < 16) ? 0 : 8;
  const int nlo   = lane & 15;
  const int mbase = (lane < 16) ? 0 : 8;
  const v8f vzero = {0.f, 0.f, 0.f, 0.f, 0.f, 0.f, 0.f, 0.f};

  if (wave < 6) {
    // -------- sc | g : [32x4096]@[4096x96] + [32x1024]@[1024x96] ----------
    const int nt = wave;
    v8f acc0 = vzero, acc1 = vzero;
    const bf16_t* r1a = &x1s[m0 * P64];
    const bf16_t* r1b = &x1s[m1 * P64];
    v16bf aL0 = load_a(&x2s[m0 * P64], 0, klo);
    v16bf aH0 = load_a(&x2s[m0 * P64], 32, klo);
    v16bf aL1 = load_a(&x2s[m1 * P64], 0, klo);
    v16bf aH1 = load_a(&x2s[m1 * P64], 32, klo);
    #pragma unroll 4
    for (int u = 0; u < 64; ++u) {
      bf16_t s0 = r1a[u], s1 = r1b[u];
      v16bf B0 = load_b(wp, FR_SS + (u * 2 + 0) * 6 + nt, lane);
      acc0 = wmma_bf16(aL0 * s0, B0, acc0);
      acc1 = wmma_bf16(aL1 * s1, B0, acc1);
      v16bf B1 = load_b(wp, FR_SS + (u * 2 + 1) * 6 + nt, lane);
      acc0 = wmma_bf16(aH0 * s0, B1, acc0);
      acc1 = wmma_bf16(aH1 * s1, B1, acc1);
    }
    v16bf c00 = load_a(&x2v[0][m0 * P32], 0, klo);
    v16bf c01 = load_a(&x2v[1][m0 * P32], 0, klo);
    v16bf c02 = load_a(&x2v[2][m0 * P32], 0, klo);
    v16bf c10 = load_a(&x2v[0][m1 * P32], 0, klo);
    v16bf c11 = load_a(&x2v[1][m1 * P32], 0, klo);
    v16bf c12 = load_a(&x2v[2][m1 * P32], 0, klo);
    const bf16_t* p00 = &x1v[0][m0 * P32];
    const bf16_t* p01 = &x1v[1][m0 * P32];
    const bf16_t* p02 = &x1v[2][m0 * P32];
    const bf16_t* p10 = &x1v[0][m1 * P32];
    const bf16_t* p11 = &x1v[1][m1 * P32];
    const bf16_t* p12 = &x1v[2][m1 * P32];
    #pragma unroll 2
    for (int u = 0; u < 32; ++u) {
      v16bf a0 = c00 * p00[u] + c01 * p01[u] + c02 * p02[u];
      v16bf a1 = c10 * p10[u] + c11 * p11[u] + c12 * p12[u];
      v16bf B  = load_b(wp, FR_VV + u * 6 + nt, lane);
      acc0 = wmma_bf16(a0, B, acc0);
      acc1 = wmma_bf16(a1, B, acc1);
    }
    #pragma unroll
    for (int r = 0; r < 8; ++r) {
      scg[(mbase + r) * PSC + nt * 16 + nlo]      = acc0[r];
      scg[(16 + mbase + r) * PSC + nt * 16 + nlo] = acc1[r];
    }

  } else if (wave < 12) {
    // -------- vec[i] : sv + vs, [32x2048]@[2048x32] twice ------------------
    const int idx = wave - 6, i = idx >> 1, nt = idx & 1;
    v8f acc0 = vzero, acc1 = vzero;
    v16bf av0 = load_a(&x2v[i][m0 * P32], 0, klo);
    v16bf av1 = load_a(&x2v[i][m1 * P32], 0, klo);
    const bf16_t* r1a = &x1s[m0 * P64];
    const bf16_t* r1b = &x1s[m1 * P64];
    #pragma unroll 4
    for (int u = 0; u < 64; ++u) {
      v16bf B = load_b(wp, FR_SV + u * 2 + nt, lane);
      acc0 = wmma_bf16(av0 * r1a[u], B, acc0);
      acc1 = wmma_bf16(av1 * r1b[u], B, acc1);
    }
    v16bf sL0 = load_a(&x2s[m0 * P64], 0, klo);
    v16bf sH0 = load_a(&x2s[m0 * P64], 32, klo);
    v16bf sL1 = load_a(&x2s[m1 * P64], 0, klo);
    v16bf sH1 = load_a(&x2s[m1 * P64], 32, klo);
    const bf16_t* pv0 = &x1v[i][m0 * P32];
    const bf16_t* pv1 = &x1v[i][m1 * P32];
    #pragma unroll 2
    for (int u = 0; u < 32; ++u) {
      bf16_t s0 = pv0[u], s1 = pv1[u];
      v16bf B0 = load_b(wp, FR_VS + (u * 2 + 0) * 2 + nt, lane);
      acc0 = wmma_bf16(sL0 * s0, B0, acc0);
      acc1 = wmma_bf16(sL1 * s1, B0, acc1);
      v16bf B1 = load_b(wp, FR_VS + (u * 2 + 1) * 2 + nt, lane);
      acc0 = wmma_bf16(sH0 * s0, B1, acc0);
      acc1 = wmma_bf16(sH1 * s1, B1, acc1);
    }
    #pragma unroll
    for (int r = 0; r < 8; ++r) {
      vecc[i][(mbase + r) * PVC + nt * 16 + nlo]      = acc0[r];
      vecc[i][(16 + mbase + r) * PVC + nt * 16 + nlo] = acc1[r];
    }

  } else if (wave == 12) {
    // -------- MLP: 128 -> 64 -> 64 -> 96, both M tiles ---------------------
    v8f acc0[6], acc1[6];
    // layer 1
    #pragma unroll
    for (int n = 0; n < 4; ++n) { acc0[n] = vzero; acc1[n] = vzero; }
    #pragma unroll
    for (int kt = 0; kt < 4; ++kt) {
      v16bf a0 = load_a(&fw[m0 * PFW], kt * 32, klo);
      v16bf a1 = load_a(&fw[m1 * PFW], kt * 32, klo);
      #pragma unroll
      for (int n = 0; n < 4; ++n) {
        v16bf B = load_b(wp, FR_FC1 + kt * 4 + n, lane);
        acc0[n] = wmma_bf16(a0, B, acc0[n]);
        acc1[n] = wmma_bf16(a1, B, acc1[n]);
      }
    }
    #pragma unroll
    for (int n = 0; n < 4; ++n)
      #pragma unroll
      for (int r = 0; r < 8; ++r) {
        float b = fcb1[n * 16 + nlo];
        hb[(mbase + r) * PH + n * 16 + nlo]        = (bf16_t)silu_f(acc0[n][r] + b);
        hb[(16 + mbase + r) * PH + n * 16 + nlo]   = (bf16_t)silu_f(acc1[n][r] + b);
      }
    asm volatile("s_wait_dscnt 0" ::: "memory");
    // layer 2
    #pragma unroll
    for (int n = 0; n < 4; ++n) { acc0[n] = vzero; acc1[n] = vzero; }
    #pragma unroll
    for (int kt = 0; kt < 2; ++kt) {
      v16bf a0 = load_a(&hb[m0 * PH], kt * 32, klo);
      v16bf a1 = load_a(&hb[m1 * PH], kt * 32, klo);
      #pragma unroll
      for (int n = 0; n < 4; ++n) {
        v16bf B = load_b(wp, FR_FC2 + kt * 4 + n, lane);
        acc0[n] = wmma_bf16(a0, B, acc0[n]);
        acc1[n] = wmma_bf16(a1, B, acc1[n]);
      }
    }
    #pragma unroll
    for (int n = 0; n < 4; ++n)
      #pragma unroll
      for (int r = 0; r < 8; ++r) {
        float b = fcb2[n * 16 + nlo];
        hb[(mbase + r) * PH + n * 16 + nlo]        = (bf16_t)silu_f(acc0[n][r] + b);
        hb[(16 + mbase + r) * PH + n * 16 + nlo]   = (bf16_t)silu_f(acc1[n][r] + b);
      }
    asm volatile("s_wait_dscnt 0" ::: "memory");
    // layer 3 (N=96)
    #pragma unroll
    for (int n = 0; n < 6; ++n) { acc0[n] = vzero; acc1[n] = vzero; }
    #pragma unroll
    for (int kt = 0; kt < 2; ++kt) {
      v16bf a0 = load_a(&hb[m0 * PH], kt * 32, klo);
      v16bf a1 = load_a(&hb[m1 * PH], kt * 32, klo);
      #pragma unroll
      for (int n = 0; n < 6; ++n) {
        v16bf B = load_b(wp, FR_FC3 + kt * 6 + n, lane);
        acc0[n] = wmma_bf16(a0, B, acc0[n]);
        acc1[n] = wmma_bf16(a1, B, acc1[n]);
      }
    }
    #pragma unroll
    for (int n = 0; n < 6; ++n)
      #pragma unroll
      for (int r = 0; r < 8; ++r) {
        float b = fcb3[n * 16 + nlo];
        wb[(mbase + r) * PW + n * 16 + nlo]      = acc0[n][r] + b;
        wb[(16 + mbase + r) * PW + n * 16 + nlo] = acc1[n][r] + b;
      }
  }
  __syncthreads();

  // ---- epilogue: gating + MLP modulation, interleave output --------------
  const float a_sc  = 0.013975424859373685f;  // 1/sqrt(S*S + V*V)
  const float a_vec = 0.015625f;              // 1/sqrt(2*S*V) = 1/64
  for (int idx = tid; idx < MB * ROW_IN; idx += 512) {
    int m = idx / ROW_IN, c = idx % ROW_IN;
    float val;
    if (c < 64) {
      float x = a_sc * scg[m * PSC + c];
      val = silu_f(x) * wb[m * PW + c];
    } else {
      int c2 = c - 64, wv = c2 / 3, i = c2 % 3;
      float g    = a_sc * scg[m * PSC + 64 + wv];
      float gate = 1.0f / (1.0f + __expf(-g));
      val = a_vec * vecc[i][m * PVC + wv] * gate * wb[m * PW + 64 + wv];
    }
    out[(size_t)(ebase + m) * ROW_IN + c] = val;
  }
}

// ---------------------------------------------------------------------------
extern "C" void kernel_launch(void* const* d_in, const int* in_sizes, int n_in,
                              void* d_out, int out_size, void* d_ws, size_t ws_size,
                              hipStream_t stream) {
  const float* fea1  = (const float*)d_in[0];
  const float* fea2  = (const float*)d_in[1];
  const float* feaw  = (const float*)d_in[2];
  const float* wss_s = (const float*)d_in[3];
  const float* wvv_s = (const float*)d_in[4];
  const float* wss_g = (const float*)d_in[5];
  const float* wvv_g = (const float*)d_in[6];
  const float* wsv   = (const float*)d_in[7];
  const float* wvs   = (const float*)d_in[8];
  const float* fc1   = (const float*)d_in[9];
  const float* fb1   = (const float*)d_in[10];
  const float* fc2   = (const float*)d_in[11];
  const float* fb2   = (const float*)d_in[12];
  const float* fc3   = (const float*)d_in[13];
  const float* fb3   = (const float*)d_in[14];
  bf16_t* wp = (bf16_t*)d_ws;   // needs FR_TOT*512*2 = ~1.25 MB

  const int packWork = FR_TOT * 32;
  equiconv_prepack<<<(packWork + 255) / 256, 256, 0, stream>>>(
      wss_s, wvv_s, wss_g, wvv_g, wsv, wvs, fc1, fc2, fc3, wp);

  equiconv_main<<<E_TOT / MB, 512, 0, stream>>>(
      fea1, fea2, feaw, fb1, fb2, fb3, wp, (float*)d_out);
}
